// Attention_Head_41291815584337
// MI455X (gfx1250) — matrix operand order repaired and checked by reading, not compiled, not run on hardware
//
#include <hip/hip_runtime.h>
#include <hip/hip_bf16.h>
#include <math.h>

// ---- problem constants (from reference) ----
#define B_  4
#define S_  2048
#define E_  1152
#define H_  12
#define D_  96
#define HD_ 1152          // H*D
#define M_  (B_*S_)       // 8192 rows

typedef __attribute__((ext_vector_type(16))) __bf16 v16bf;
typedef __attribute__((ext_vector_type(8)))  float  v8f;

// ---------- helpers ----------
__device__ __forceinline__ unsigned short f2bf(float f) {
    unsigned u = __float_as_uint(f);
    u += 0x7FFFu + ((u >> 16) & 1u);      // round-to-nearest-even
    return (unsigned short)(u >> 16);
}

// 16 contiguous bf16 (B-tile / column-contiguous)
__device__ __forceinline__ v16bf ld16(const unsigned short* p) {
    union { uint4 u[2]; v16bf v; } c;
    c.u[0] = *(const uint4*)(p);
    c.u[1] = *(const uint4*)(p + 8);
    return c.v;
}
// A-tile fragment: 16 contiguous bf16 at p (8 at p, 8 at p+8)
__device__ __forceinline__ v16bf ld8x2(const unsigned short* p) {
    union { uint4 u[2]; v16bf v; } c;
    c.u[0] = *(const uint4*)(p);
    c.u[1] = *(const uint4*)(p + 8);
    return c.v;
}

__device__ __forceinline__ v8f wmma_bf16(v16bf a, v16bf b, v8f c) {
    return __builtin_amdgcn_wmma_f32_16x16x32_bf16(false, a, false, b, (short)0, c, false, false);
}

// CDNA5 async copy: global -> LDS, 16 bytes per lane, tracked by ASYNCcnt.
// lds_off = wave-relative LDS byte offset (flat-addr low 32 bits), gaddr = global byte address.
__device__ __forceinline__ void async_b128(unsigned lds_off, const void* gp) {
    unsigned long long ga = (unsigned long long)gp;
    asm volatile("global_load_async_to_lds_b128 %0, %1, off"
                 :: "v"(lds_off), "v"(ga) : "memory");
}
__device__ __forceinline__ void wait_async0() {
    asm volatile("s_wait_asynccnt 0x0" ::: "memory");
}
__device__ __forceinline__ unsigned lds_off_of(const void* p) {
    return (unsigned)(unsigned long long)p;   // LDS flat addr low 32 bits = LDS offset
}

// ---------- prep: fp32 -> bf16 ----------
__global__ void cvt_bf16(const float* __restrict__ in, unsigned short* __restrict__ out, int n) {
    int i = blockIdx.x * blockDim.x + threadIdx.x;
    if (i < n) out[i] = f2bf(in[i]);
}

// in: rows x cols fp32 (row-major)  ->  out: cols x rows bf16 (row-major)
__global__ void transp_bf16(const float* __restrict__ in, unsigned short* __restrict__ out,
                            int rows, int cols) {
    int i = blockIdx.x * blockDim.x + threadIdx.x;
    if (i >= rows * cols) return;
    int r = i / cols, c = i % cols;
    out[(size_t)c * rows + r] = f2bf(in[i]);
}

// ---------- fused QKV projection + bias + RoPE ----------
// grid: (M/128, HD/64, 3)  block: 256 (8 waves; wave w computes rows m0+16w..+15, 64 cols)
// B tile (64 cols x 32 k = 4 KB) staged in LDS via async copy, double-buffered.
__global__ __launch_bounds__(256)
void qkv_rope(const unsigned short* __restrict__ X,     // M x E bf16
              const unsigned short* __restrict__ WqT,   // HD x E bf16 (transposed)
              const unsigned short* __restrict__ WkT,
              const unsigned short* __restrict__ WvT,
              const float* __restrict__ bq, const float* __restrict__ bk,
              const float* __restrict__ bv,
              unsigned short* __restrict__ qo,          // (b,h,s,d) bf16
              unsigned short* __restrict__ ko,          // (b,h,s,d) bf16
              unsigned short* __restrict__ vTo)         // (b,h,d,s) bf16
{
    __shared__ unsigned short bsm[2][64][40];           // 64 B-columns x 32 k, 80B padded rows

    const int tid  = threadIdx.x;
    const int wav  = tid >> 5;
    const int lane = tid & 31;
    const int half = (lane >> 4) & 1;
    const int ln   = lane & 15;
    const int m0   = blockIdx.x * 128 + wav * 16;
    const int n0   = blockIdx.y * 64;
    const int z    = blockIdx.z;

    const unsigned short* WT  = (z == 0) ? WqT : (z == 1) ? WkT : WvT;
    const float*          bia = (z == 0) ? bq  : (z == 1) ? bk  : bv;

    // async staging coords: thread loads 16B: B-column row srow, k-chunk schk
    const int srow = tid >> 2;              // 0..63
    const int schk = (tid & 3) * 8;         // ushort offset within 32-k row
    const unsigned short* sptr = WT + (size_t)(n0 + srow) * E_ + schk;

    // prologue: stage k-step 0, load A fragment 0
    async_b128(lds_off_of(&bsm[0][srow][schk]), sptr);
    const unsigned short* arow = X + (size_t)(m0 + ln) * E_ + half * 16;
    v16bf a = ld8x2(arow);

    const v8f zacc = {0.f,0.f,0.f,0.f,0.f,0.f,0.f,0.f};
    v8f acc[4];
#pragma unroll
    for (int t = 0; t < 4; ++t) acc[t] = zacc;

    const int NK = E_ / 32;                 // 36 k-steps
    for (int ks = 0; ks < NK; ++ks) {
        const int buf = ks & 1;
        wait_async0();                      // my stage(ks) ops landed
        __syncthreads();                    // everyone's stage(ks) landed
        v16bf b0 = ld16(&bsm[buf][ 0 + ln][half * 16]);
        v16bf b1 = ld16(&bsm[buf][16 + ln][half * 16]);
        v16bf b2 = ld16(&bsm[buf][32 + ln][half * 16]);
        v16bf b3 = ld16(&bsm[buf][48 + ln][half * 16]);
        v16bf an = a;
        if (ks + 1 < NK) {                  // overlap next stage + A load with WMMAs
            async_b128(lds_off_of(&bsm[buf ^ 1][srow][schk]), sptr + (ks + 1) * 32);
            an = ld8x2(arow + (ks + 1) * 32);
        }
        acc[0] = wmma_bf16(a, b0, acc[0]);
        acc[1] = wmma_bf16(a, b1, acc[1]);
        acc[2] = wmma_bf16(a, b2, acc[2]);
        acc[3] = wmma_bf16(a, b3, acc[3]);
        a = an;
    }

    const bool doRope = (z < 2);
#pragma unroll
    for (int t = 0; t < 4; ++t) {
        const int   col = n0 + t*16 + ln;
        const float bb  = bia[col];
        const int   h   = col / D_;
        const int   dl  = col % D_;
        const float theta = __powf(10000.f, -(float)(dl & ~1) / (float)D_);
        const bool  even  = (dl & 1) == 0;
#pragma unroll
        for (int j = 0; j < 8; ++j) {
            const int row = m0 + j + half * 8;
            const int b   = row >> 11;           // row / S_
            const int s   = row & (S_ - 1);
            float v = acc[t][j] + bb;
            if (doRope) {
                float pv = __shfl_xor(v, 1);     // partner column (dl ^ 1)
                float sn, cs;
                __sincosf((float)s * theta, &sn, &cs);
                v = even ? (v * cs - pv * sn) : (v * cs + pv * sn);
            }
            const unsigned short bfv = f2bf(v);
            if (z == 0)      qo [((size_t)(b*H_ + h)*S_ + s)*D_ + dl] = bfv;
            else if (z == 1) ko [((size_t)(b*H_ + h)*S_ + s)*D_ + dl] = bfv;
            else             vTo[((size_t)(b*H_ + h)*D_ + dl)*S_ + s] = bfv;
        }
    }
}

// ---------- causal flash attention ----------
// grid: (S/16, H, B)  block: 32 (one wave handles a 16-query tile)
__global__ __launch_bounds__(32)
void attn(const unsigned short* __restrict__ qb,   // (b,h,s,d) bf16 (RoPE'd)
          const unsigned short* __restrict__ kb,   // (b,h,s,d) bf16 (RoPE'd)
          const unsigned short* __restrict__ vtb,  // (b,h,d,s) bf16
          unsigned short* __restrict__ ctx)        // (b,s,h*d) bf16
{
    __shared__ unsigned short plds[16][40];        // P tile 16x32 bf16, padded rows

    const int lane = threadIdx.x;
    const int half = lane >> 4;
    const int ln   = lane & 15;
    const int m0   = blockIdx.x * 16;
    const int h    = blockIdx.y;
    const int b    = blockIdx.z;

    const size_t bh = (size_t)b * H_ + h;
    const unsigned short* Q  = qb  + bh * S_ * D_;
    const unsigned short* K  = kb  + bh * S_ * D_;
    const unsigned short* VT = vtb + bh * (size_t)D_ * S_;

    // Q tile in A layout: 3 k-steps cover d=0..95
    v16bf qa[3];
#pragma unroll
    for (int kk = 0; kk < 3; ++kk)
        qa[kk] = ld8x2(Q + (size_t)(m0 + ln) * D_ + kk * 32 + half * 16);

    const float scale = 0.10206207261596575f;      // 1/sqrt(96)
    const v8f zacc = {0.f,0.f,0.f,0.f,0.f,0.f,0.f,0.f};

    float mrow[8], lrow[8];
#pragma unroll
    for (int j = 0; j < 8; ++j) { mrow[j] = -INFINITY; lrow[j] = 0.f; }
    v8f oacc[6];
#pragma unroll
    for (int t = 0; t < 6; ++t) oacc[t] = zacc;

    const int kbEnd = (m0 + 16 + 31) >> 5;         // key blocks of 32 needed for causality
    for (int kbi = 0; kbi < kbEnd; ++kbi) {
        const int key0 = kbi * 32;

        // load all 6 K B-tiles first (loads clause together), then 6 back-to-back WMMAs
        v16bf kb0[3], kb1[3];
#pragma unroll
        for (int kk = 0; kk < 3; ++kk) {
            kb0[kk] = ld16(K + (size_t)(key0      + ln) * D_ + kk * 32 + half * 16);
            kb1[kk] = ld16(K + (size_t)(key0 + 16 + ln) * D_ + kk * 32 + half * 16);
        }
        v8f s0 = zacc, s1 = zacc;
#pragma unroll
        for (int kk = 0; kk < 3; ++kk) s0 = wmma_bf16(qa[kk], kb0[kk], s0);
#pragma unroll
        for (int kk = 0; kk < 3; ++kk) s1 = wmma_bf16(qa[kk], kb1[kk], s1);

        // online softmax (rows of C tile live in 16-lane halves)
        float corr8[8];
#pragma unroll
        for (int j = 0; j < 8; ++j) {
            const int qrow = m0 + j + half * 8;
            float v0 = s0[j] * scale;
            float v1 = s1[j] * scale;
            if (key0 + ln      > qrow) v0 = -INFINITY;
            if (key0 + 16 + ln > qrow) v1 = -INFINITY;
            float mx = fmaxf(v0, v1);
#pragma unroll
            for (int msk = 8; msk >= 1; msk >>= 1) mx = fmaxf(mx, __shfl_xor(mx, msk));
            const float nm   = fmaxf(mrow[j], mx);
            const float corr = __expf(mrow[j] - nm);
            const float p0 = __expf(v0 - nm);
            const float p1 = __expf(v1 - nm);
            float rs = p0 + p1;
#pragma unroll
            for (int msk = 8; msk >= 1; msk >>= 1) rs += __shfl_xor(rs, msk);
            lrow[j] = lrow[j] * corr + rs;
            mrow[j] = nm;
            corr8[j] = corr;
            const int prow = j + half * 8;
            plds[prow][ln]      = f2bf(p0);
            plds[prow][16 + ln] = f2bf(p1);
        }
#pragma unroll
        for (int t = 0; t < 6; ++t)
#pragma unroll
            for (int j = 0; j < 8; ++j) oacc[t][j] *= corr8[j];

        __syncthreads();
        // P tile back out of LDS in A layout
        v16bf pa = ld8x2(&plds[ln][half * 16]);
        // load all 6 V B-tiles, then 6 back-to-back WMMAs
        v16bf vb[6];
#pragma unroll
        for (int t = 0; t < 6; ++t)
            vb[t] = ld16(VT + (size_t)(t*16 + ln) * S_ + key0 + half * 16);
#pragma unroll
        for (int t = 0; t < 6; ++t) oacc[t] = wmma_bf16(pa, vb[t], oacc[t]);
        __syncthreads();
    }

    // normalize + store ctx (b, s, h*d) bf16
#pragma unroll
    for (int j = 0; j < 8; ++j) {
        const float inv = 1.f / lrow[j];
        const int   s   = m0 + j + half * 8;
#pragma unroll
        for (int t = 0; t < 6; ++t) {
            const int d = t*16 + ln;
            ctx[((size_t)b * S_ + s) * HD_ + h * D_ + d] = f2bf(oacc[t][j] * inv);
        }
    }
}

// ---------- output projection: out = ctx @ Wo + bo (fp32 out) ----------
// grid: (M/128, E/64)  block: 256 (8 waves), LDS-staged B with async copy
__global__ __launch_bounds__(256)
void out_proj(const unsigned short* __restrict__ Xc,   // M x HD bf16
              const unsigned short* __restrict__ WoT,  // E x HD bf16 (transposed)
              const float* __restrict__ bo,
              float* __restrict__ out)                 // M x E fp32
{
    __shared__ unsigned short bsm[2][64][40];

    const int tid  = threadIdx.x;
    const int wav  = tid >> 5;
    const int lane = tid & 31;
    const int half = (lane >> 4) & 1;
    const int ln   = lane & 15;
    const int m0   = blockIdx.x * 128 + wav * 16;
    const int n0   = blockIdx.y * 64;

    const int srow = tid >> 2;
    const int schk = (tid & 3) * 8;
    const unsigned short* sptr = WoT + (size_t)(n0 + srow) * HD_ + schk;

    async_b128(lds_off_of(&bsm[0][srow][schk]), sptr);
    const unsigned short* arow = Xc + (size_t)(m0 + ln) * HD_ + half * 16;
    v16bf a = ld8x2(arow);

    const v8f zacc = {0.f,0.f,0.f,0.f,0.f,0.f,0.f,0.f};
    v8f acc[4];
#pragma unroll
    for (int t = 0; t < 4; ++t) acc[t] = zacc;

    const int NK = HD_ / 32;
    for (int ks = 0; ks < NK; ++ks) {
        const int buf = ks & 1;
        wait_async0();
        __syncthreads();
        v16bf b0 = ld16(&bsm[buf][ 0 + ln][half * 16]);
        v16bf b1 = ld16(&bsm[buf][16 + ln][half * 16]);
        v16bf b2 = ld16(&bsm[buf][32 + ln][half * 16]);
        v16bf b3 = ld16(&bsm[buf][48 + ln][half * 16]);
        v16bf an = a;
        if (ks + 1 < NK) {
            async_b128(lds_off_of(&bsm[buf ^ 1][srow][schk]), sptr + (ks + 1) * 32);
            an = ld8x2(arow + (ks + 1) * 32);
        }
        acc[0] = wmma_bf16(a, b0, acc[0]);
        acc[1] = wmma_bf16(a, b1, acc[1]);
        acc[2] = wmma_bf16(a, b2, acc[2]);
        acc[3] = wmma_bf16(a, b3, acc[3]);
        a = an;
    }

#pragma unroll
    for (int t = 0; t < 4; ++t) {
        const int   col = n0 + t*16 + ln;
        const float bb  = bo[col];
#pragma unroll
        for (int j = 0; j < 8; ++j) {
            const int row = m0 + j + half * 8;
            out[(size_t)row * E_ + col] = acc[t][j] + bb;
        }
    }
}

// ---------- launch ----------
extern "C" void kernel_launch(void* const* d_in, const int* in_sizes, int n_in,
                              void* d_out, int out_size, void* d_ws, size_t ws_size,
                              hipStream_t stream) {
    (void)in_sizes; (void)n_in; (void)out_size; (void)ws_size;
    const float* logits = (const float*)d_in[0];
    const float* Wq = (const float*)d_in[1];
    const float* bq = (const float*)d_in[2];
    const float* Wk = (const float*)d_in[3];
    const float* bk = (const float*)d_in[4];
    const float* Wv = (const float*)d_in[5];
    const float* bv = (const float*)d_in[6];
    const float* Wo = (const float*)d_in[7];
    const float* bo = (const float*)d_in[8];
    float* out = (float*)d_out;

    unsigned short* ws = (unsigned short*)d_ws;
    const size_t SZQ = (size_t)B_ * H_ * S_ * D_;   // per q/k/v element count
    unsigned short* qbf   = ws;
    unsigned short* kbf   = qbf   + SZQ;
    unsigned short* vtbf  = kbf   + SZQ;
    unsigned short* ctxbf = vtbf  + SZQ;
    unsigned short* xbf   = ctxbf + (size_t)M_ * HD_;
    unsigned short* WqT   = xbf   + (size_t)M_ * E_;
    unsigned short* WkT   = WqT   + (size_t)HD_ * E_;
    unsigned short* WvT   = WkT   + (size_t)HD_ * E_;
    unsigned short* WoT   = WvT   + (size_t)HD_ * E_;

    const int nX = M_ * E_;
    cvt_bf16<<<(nX + 255) / 256, 256, 0, stream>>>(logits, xbf, nX);
    const int nW = E_ * HD_;
    transp_bf16<<<(nW + 255) / 256, 256, 0, stream>>>(Wq, WqT, E_,  HD_);
    transp_bf16<<<(nW + 255) / 256, 256, 0, stream>>>(Wk, WkT, E_,  HD_);
    transp_bf16<<<(nW + 255) / 256, 256, 0, stream>>>(Wv, WvT, E_,  HD_);
    transp_bf16<<<(nW + 255) / 256, 256, 0, stream>>>(Wo, WoT, HD_, E_);

    dim3 gq(M_ / 128, HD_ / 64, 3);
    qkv_rope<<<gq, 256, 0, stream>>>(xbf, WqT, WkT, WvT, bq, bk, bv, qbf, kbf, vtbf);

    dim3 ga(S_ / 16, H_, B_);
    attn<<<ga, 32, 0, stream>>>(qbf, kbf, vtbf, ctxbf);

    dim3 go(M_ / 128, E_ / 64);
    out_proj<<<go, 256, 0, stream>>>(ctxbf, WoT, bo, out);
}
